// LaneAwareProbabilityLearning_50405736186268
// MI455X (gfx1250) — matrix operand order repaired and checked by reading, not compiled, not run on hardware
//
#include <hip/hip_runtime.h>
#include <hip/hip_bf16.h>

// ---- model constants (match reference) ----
#define H_DIM   128      // hidden
#define DI_DIM  256      // d_inner
#define N_STATE 16       // d_state
#define DTR_DIM 8        // dt_rank
#define KC_DIM  4        // conv kernel
#define LANE_D  64
#define NLANES  6
#define NB      4
#define B_SZ    8
#define L_SEQ   2048
#define M_TOK   (B_SZ * L_SEQ)     // 16384 token rows
#define XZ_W    (2 * DI_DIM)       // 512
#define XD_W    48                 // 40 padded to 48
#define LO_PAD  16                 // 6 padded to 16

typedef __attribute__((ext_vector_type(16))) _Float16 v16h;
typedef __attribute__((ext_vector_type(8)))  _Float16 v8h;
typedef __attribute__((ext_vector_type(8)))  float    v8f;

// ---------------- f32 -> f16 convert (activations, no pad) ----------------
__global__ void cvt_f32_f16(const float* __restrict__ src, _Float16* __restrict__ dst, int total) {
    int idx = blockIdx.x * blockDim.x + threadIdx.x;
    if (idx >= total) return;
    dst[idx] = (_Float16)src[idx];
}

// ---------------- pack weight B[K,Ncols] (f32, row-major) into WMMA fragment layout ----
// dst layout: [(tn*kBlocks + kb)*32 + lane]*16 + j  where
//   half = lane>>4, r = lane&15, col = tn*16 + r, k = kb*32 + (j<8 ? 8*half + j : 16 + 8*half + (j-8))
// (matches A fill: lanes 0-15 hold K 0..7,16..23; lanes 16-31 hold K 8..15,24..31)
__global__ void pack_b_wmma(const float* __restrict__ src, _Float16* __restrict__ dst,
                            int K, int Ncols, int tilesN) {
    int idx = blockIdx.x * blockDim.x + threadIdx.x;
    int kBlocks = K >> 5;
    int total = tilesN * kBlocks * 32 * 16;
    if (idx >= total) return;
    int j    = idx & 15;
    int lane = (idx >> 4) & 31;
    int rest = idx >> 9;
    int kb = rest % kBlocks;
    int tn = rest / kBlocks;
    int half = lane >> 4, r = lane & 15;
    int col = tn * 16 + r;
    int k   = kb * 32 + ((j < 8) ? (8 * half + j) : (16 + 8 * half + (j - 8)));
    dst[idx] = (col < Ncols) ? (_Float16)src[k * Ncols + col] : (_Float16)0.0f;
}

// ---------------- RMSNorm (one token per 128-thread block) ----------------
__global__ void rmsnorm_f16(const float* __restrict__ x, const float* __restrict__ w,
                            _Float16* __restrict__ out) {
    __shared__ float red[H_DIM];
    int t = blockIdx.x, i = threadIdx.x;
    float v = x[t * H_DIM + i];
    red[i] = v * v;
    __syncthreads();
    for (int s = H_DIM / 2; s > 0; s >>= 1) {
        if (i < s) red[i] += red[i + s];
        __syncthreads();
    }
    float scale = rsqrtf(red[0] * (1.0f / H_DIM) + 1e-5f);
    out[t * H_DIM + i] = (_Float16)(v * scale * w[i]);
}

// ---------------- WMMA GEMM: C[M,*] = A[M,K](f16) * Bpacked (+bias,+res) ----------------
// One 16x16 output tile per wave32; K multiple of 32.  B is pre-packed in fragment order:
// inner loop = 2x b128 (A) + 2x b128 (B) + 1x v_wmma_f32_16x16x32_f16.
__global__ __launch_bounds__(256)
void gemm_wmma_f16(const _Float16* __restrict__ A, const _Float16* __restrict__ Bp,
                   const float* __restrict__ bias, const float* __restrict__ res,
                   float* __restrict__ C, _Float16* __restrict__ C16,
                   int M, int K, int Npad, int Nvalid, int ldOut) {
    int gtid   = blockIdx.x * blockDim.x + threadIdx.x;
    int wave   = gtid >> 5;
    int lane   = gtid & 31;
    int tilesN = Npad >> 4;
    int kBlocks = K >> 5;
    int tm = wave / tilesN;
    int tn = wave - tm * tilesN;
    if (tm * 16 >= M) return;               // wave-uniform guard

    int r = lane & 15, half = lane >> 4;
    const _Float16* Arow = A + (tm * 16 + r) * K + half * 8;
    const _Float16* Bfrag = Bp + (tn * kBlocks * 32 + lane) * 16;
    int colN = tn * 16 + r;

    v8f acc = {};
    for (int kb = 0; kb < kBlocks; ++kb) {
        v8h alo = *(const v8h*)(Arow);          // K[k0 + 8*half + 0..7]
        v8h ahi = *(const v8h*)(Arow + 16);     // K[k0 + 16 + 8*half + 0..7]
        Arow += 32;
        v8h blo = *(const v8h*)(Bfrag);
        v8h bhi = *(const v8h*)(Bfrag + 8);
        Bfrag += 32 * 16;
        v16h a, b;
#pragma unroll
        for (int j = 0; j < 8; ++j) {
            a[j] = alo[j]; a[j + 8] = ahi[j];
            b[j] = blo[j]; b[j + 8] = bhi[j];
        }
        acc = __builtin_amdgcn_wmma_f32_16x16x32_f16(
            /*neg_a=*/false, a, /*neg_b=*/false, b,
            /*c_mod=*/(short)0, acc, /*reuse_a=*/false, /*reuse_b=*/false);
    }
#pragma unroll
    for (int v = 0; v < 8; ++v) {
        int row = tm * 16 + half * 8 + v;    // C/D layout: VGPR v -> M = 8*half + v
        if (colN < Nvalid) {
            float val = acc[v];
            if (bias) val += bias[colN];
            if (res)  val += res[row * ldOut + colN];
            C[row * ldOut + colN] = val;
            if (C16) C16[row * Npad + colN] = (_Float16)val;
        }
    }
}

// ---------------- depthwise causal conv1d (k=4) + SiLU; reads xc = xz[:, 0:256] ----------------
__global__ void conv_silu(const float* __restrict__ xz, const float* __restrict__ cw,
                          const float* __restrict__ cb, float* __restrict__ u,
                          _Float16* __restrict__ u16) {
    int idx = blockIdx.x * blockDim.x + threadIdx.x;
    if (idx >= M_TOK * DI_DIM) return;
    int d = idx & (DI_DIM - 1);
    int tok = idx / DI_DIM;
    int l = tok & (L_SEQ - 1);
    float acc = cb[d];
#pragma unroll
    for (int j = 0; j < KC_DIM; ++j) {
        int back = (KC_DIM - 1) - j;        // taps x[l - back]
        if (l >= back)
            acc += cw[d * KC_DIM + j] * xz[(tok - back) * XZ_W + d];
    }
    float s = acc / (1.0f + __expf(-acc));   // silu
    u[idx]   = s;
    u16[idx] = (_Float16)s;
}

// ---------------- dt = softplus(dt_r @ W_dt + b_dt)   (K=8 -> scalar) ----------------
__global__ void dt_proj(const float* __restrict__ xdbl, const float* __restrict__ Wdt,
                        const float* __restrict__ bdt, float* __restrict__ dt) {
    int idx = blockIdx.x * blockDim.x + threadIdx.x;
    if (idx >= M_TOK * DI_DIM) return;
    int d = idx & (DI_DIM - 1);
    int tok = idx / DI_DIM;
    const float* dtr = xdbl + tok * XD_W;
    float acc = bdt[d];
#pragma unroll
    for (int rr = 0; rr < DTR_DIM; ++rr) acc += dtr[rr] * Wdt[rr * DI_DIM + d];
    dt[idx] = (acc > 20.0f) ? acc : log1pf(__expf(acc));
}

// ---------------- fused selective scan + skip(D) + SiLU(z) gate -> y16 ----------------
// block = one batch (256 threads = one d-channel each), sequential over L; h[16] in VGPRs.
__global__ __launch_bounds__(DI_DIM)
void ssm_scan(const float* __restrict__ dt, const float* __restrict__ u,
              const float* __restrict__ xdbl, const float* __restrict__ xz,
              const float* __restrict__ A_log, const float* __restrict__ Dw,
              _Float16* __restrict__ y16) {
    int b = blockIdx.x;
    int d = threadIdx.x;
    float An[N_STATE];
#pragma unroll
    for (int n = 0; n < N_STATE; ++n) An[n] = -__expf(A_log[d * N_STATE + n]);
    float Dd = Dw[d];
    float h[N_STATE];
#pragma unroll
    for (int n = 0; n < N_STATE; ++n) h[n] = 0.0f;

    for (int l = 0; l < L_SEQ; ++l) {
        int tok = b * L_SEQ + l;
        float dtv = dt[tok * DI_DIM + d];
        float uv  = u[tok * DI_DIM + d];
        float zv  = xz[tok * XZ_W + DI_DIM + d];
        const float* bc = xdbl + tok * XD_W;     // block-uniform address -> scalar loads
        float du = dtv * uv;
        float y = 0.0f;
#pragma unroll
        for (int n = 0; n < N_STATE; ++n) {
            float dA = __expf(dtv * An[n]);
            h[n] = dA * h[n] + du * bc[DTR_DIM + n];            // Bm
            y   += h[n] * bc[DTR_DIM + N_STATE + n];            // Cm
        }
        y += uv * Dd;
        float g = zv / (1.0f + __expf(-zv));                    // silu(z)
        y16[tok * DI_DIM + d] = (_Float16)(y * g);
    }
}

// =====================================================================================
extern "C" void kernel_launch(void* const* d_in, const int* in_sizes, int n_in,
                              void* d_out, int out_size, void* d_ws, size_t ws_size,
                              hipStream_t stream) {
    (void)in_sizes; (void)n_in; (void)out_size; (void)ws_size;
    // input order: agent_features(unused), lane_features, W_proj, b_proj,
    //   [norm_w, W_in, conv_w, conv_b, W_xproj, W_dt, b_dt, A_log, D, W_out] x4,
    //   W_li, b_li, W_lo, b_lo
    const float* lane   = (const float*)d_in[1];
    const float* W_proj = (const float*)d_in[2];
    const float* b_proj = (const float*)d_in[3];
    const float* W_li   = (const float*)d_in[4 + 10 * NB + 0];
    const float* b_li   = (const float*)d_in[4 + 10 * NB + 1];
    const float* W_lo   = (const float*)d_in[4 + 10 * NB + 2];
    const float* b_lo   = (const float*)d_in[4 + 10 * NB + 3];
    auto blk = [&](int i, int j) { return (const float*)d_in[4 + 10 * i + j]; };

    // ---- bump allocator over workspace ----
    char* wp = (char*)d_ws;
    auto alloc = [&](size_t bytes) { void* p = (void*)wp; wp += (bytes + 255) & ~(size_t)255; return p; };
    float*     x      = (float*)    alloc((size_t)M_TOK * H_DIM  * 4);
    _Float16*  x16    = (_Float16*) alloc((size_t)M_TOK * H_DIM  * 2);
    _Float16*  lane16 = (_Float16*) alloc((size_t)M_TOK * LANE_D * 2);
    _Float16*  h16    = (_Float16*) alloc((size_t)M_TOK * H_DIM  * 2);
    float*     xz     = (float*)    alloc((size_t)M_TOK * XZ_W   * 4);
    float*     u      = (float*)    alloc((size_t)M_TOK * DI_DIM * 4);
    _Float16*  u16    = (_Float16*) alloc((size_t)M_TOK * DI_DIM * 2);
    float*     xdbl   = (float*)    alloc((size_t)M_TOK * XD_W   * 4);
    float*     dtb    = (float*)    alloc((size_t)M_TOK * DI_DIM * 4);
    _Float16*  y16    = (_Float16*) alloc((size_t)M_TOK * DI_DIM * 2);
    float*     tbuf   = (float*)    alloc((size_t)M_TOK * H_DIM  * 4);
    _Float16*  t16    = (_Float16*) alloc((size_t)M_TOK * H_DIM  * 2);
    _Float16*  wProj16 = (_Float16*)alloc((size_t)LANE_D * H_DIM * 2);
    _Float16*  wLi16   = (_Float16*)alloc((size_t)H_DIM * H_DIM * 2);
    _Float16*  wLo16   = (_Float16*)alloc((size_t)H_DIM * LO_PAD * 2);
    _Float16*  wIn16[NB];  _Float16* wXp16[NB];  _Float16* wOut16[NB];
    for (int i = 0; i < NB; ++i) {
        wIn16[i]  = (_Float16*)alloc((size_t)H_DIM  * XZ_W  * 2);
        wXp16[i]  = (_Float16*)alloc((size_t)DI_DIM * XD_W  * 2);
        wOut16[i] = (_Float16*)alloc((size_t)DI_DIM * H_DIM * 2);
    }

    // pack weight into WMMA-fragment order (Npad = tilesN*16, K multiple of 32)
    auto packB = [&](const float* s, _Float16* dst, int K, int Ncols, int Npad) {
        int tot = Npad * K;     // tilesN * kBlocks * 512
        pack_b_wmma<<<(tot + 255) / 256, 256, 0, stream>>>(s, dst, K, Ncols, Npad / 16);
    };
    auto gemm = [&](const _Float16* A, const _Float16* Bp, const float* bias, const float* res,
                    float* C, _Float16* C16, int M, int K, int Npad, int Nvalid, int ldOut) {
        int waves = (M / 16) * (Npad / 16);
        gemm_wmma_f16<<<(waves * 32 + 255) / 256, 256, 0, stream>>>(
            A, Bp, bias, res, C, C16, M, K, Npad, Nvalid, ldOut);
    };

    // ---- weight packing (f32 -> f16 fragment layout, zero-padded columns) ----
    {
        int tot = M_TOK * LANE_D;
        cvt_f32_f16<<<(tot + 255) / 256, 256, 0, stream>>>(lane, lane16, tot);
    }
    packB(W_proj, wProj16, LANE_D, H_DIM, H_DIM);
    packB(W_li,   wLi16,   H_DIM,  H_DIM, H_DIM);
    packB(W_lo,   wLo16,   H_DIM,  NLANES, LO_PAD);
    for (int i = 0; i < NB; ++i) {
        packB(blk(i, 1), wIn16[i],  H_DIM,  XZ_W,                   XZ_W);
        packB(blk(i, 4), wXp16[i],  DI_DIM, DTR_DIM + 2 * N_STATE,  XD_W);
        packB(blk(i, 9), wOut16[i], DI_DIM, H_DIM,                  H_DIM);
    }

    // ---- input projection: x = lane @ W_proj + b_proj ----
    gemm(lane16, wProj16, b_proj, nullptr, x, nullptr, M_TOK, LANE_D, H_DIM, H_DIM, H_DIM);

    const int elemGrid = (M_TOK * DI_DIM + 255) / 256;
    for (int i = 0; i < NB; ++i) {
        // RMSNorm -> h16
        rmsnorm_f16<<<M_TOK, H_DIM, 0, stream>>>(x, blk(i, 0), h16);
        // xz = h @ W_in
        gemm(h16, wIn16[i], nullptr, nullptr, xz, nullptr, M_TOK, H_DIM, XZ_W, XZ_W, XZ_W);
        // u = silu(conv1d(xc) + conv_b)
        conv_silu<<<elemGrid, 256, 0, stream>>>(xz, blk(i, 2), blk(i, 3), u, u16);
        // xdbl = u @ W_xproj   (dt_r | B | C, padded to 48)
        gemm(u16, wXp16[i], nullptr, nullptr, xdbl, nullptr, M_TOK, DI_DIM, XD_W, XD_W, XD_W);
        // dt = softplus(dt_r @ W_dt + b_dt)
        dt_proj<<<elemGrid, 256, 0, stream>>>(xdbl, blk(i, 5), blk(i, 6), dtb);
        // fused selective scan + D skip + silu(z) gate
        ssm_scan<<<B_SZ, DI_DIM, 0, stream>>>(dtb, u, xdbl, xz, blk(i, 7), blk(i, 8), y16);
        // x = x + y @ W_out  (in-place residual, also emit x16)
        gemm(y16, wOut16[i], nullptr, x, x, x16, M_TOK, DI_DIM, H_DIM, H_DIM, H_DIM);
    }

    // ---- head: t = x @ W_li + b_li ; out = t @ W_lo + b_lo ----
    gemm(x16, wLi16, b_li, nullptr, tbuf, t16, M_TOK, H_DIM, H_DIM, H_DIM, H_DIM);
    gemm(t16, wLo16, b_lo, nullptr, (float*)d_out, nullptr, M_TOK, H_DIM, LO_PAD, NLANES, NLANES);
}
